// SetwiseDistance_70806830842610
// MI455X (gfx1250) — matrix operand order repaired and checked by reading, not compiled, not run on hardware
//
#include <hip/hip_runtime.h>
#include <hip/hip_bf16.h>
#include <stdint.h>

// MI455X / gfx1250: wave32, WMMA 16x16x32 bf16 -> f32 accumulate.
// Fused smooth-chamfer: bf16 GEMM tiles + exp/log pooling epilogue in LDS.

typedef __attribute__((ext_vector_type(16))) __bf16 v16bf;
typedef __attribute__((ext_vector_type(8)))  float  v8f;

#define D_DIM   1024
#define N_IMG   8192
#define N_TXT   8192
#define BM      128              // dist-tile rows per workgroup
#define BN      256              // dist-tile cols per workgroup
#define BI_OUT  (N_IMG / 4)      // 2048
#define BT_OUT  (N_TXT / 4)      // 2048
#define LDS_STRIDE (BN + 4)      // pad: 8*260 % 64 != 0 -> no bank conflict

// ---------------------------------------------------------------- convert ---
__device__ __forceinline__ uint32_t pack2_bf16(float lo, float hi) {
  uint32_t l = __float_as_uint(lo);
  uint32_t h = __float_as_uint(hi);
  l = (l + 0x7FFFu + ((l >> 16) & 1u)) >> 16;           // RNE to bf16
  h = (h + 0x7FFFu + ((h >> 16) & 1u)) & 0xFFFF0000u;
  return h | l;
}

__global__ __launch_bounds__(256) void cvt_f32_bf16_kernel(
    const float* __restrict__ in, uint16_t* __restrict__ out) {
  // 8 elements per thread, fully covering N*D (multiple of 2048 threads*8)
  size_t i = ((size_t)blockIdx.x * blockDim.x + threadIdx.x) * 8;
  float4 a = *(const float4*)(in + i);
  float4 b = *(const float4*)(in + i + 4);
  uint4 p;
  p.x = pack2_bf16(a.x, a.y);
  p.y = pack2_bf16(a.z, a.w);
  p.z = pack2_bf16(b.x, b.y);
  p.w = pack2_bf16(b.z, b.w);
  *(uint4*)(out + i) = p;
}

// ------------------------------------------------------- fused GEMM + pool ---
union AFrag { v16bf v; uint4 q[2]; };

__global__ __launch_bounds__(256) void chamfer_gemm_kernel(
    const uint16_t* __restrict__ Abf,   // img bf16 [8192,1024] row-major
    const uint16_t* __restrict__ Bbf,   // txt bf16 [8192,1024] row-major
    float* __restrict__ out)            // [2048,2048]
{
  extern __shared__ float lds[];        // BM x LDS_STRIDE floats (133 KB)

  const int tid  = threadIdx.x;
  const int lane = tid & 31;
  const int wave = tid >> 5;            // 0..7
  const int wm   = wave & 1;            // 2 wave-rows  (64 rows each)
  const int wn   = wave >> 1;           // 4 wave-cols  (64 cols each)

  const int bm = blockIdx.y * BM;       // img row base
  const int bn = blockIdx.x * BN;       // txt row base

  const int lrow    = lane & 15;        // matrix row within 16x16 tile
  const int khalfA  = (lane >> 4) << 4; // A: lanes16-31 start at K=16
  const int khalfB  = (lane >> 4) << 4; // B: lanes16-31 start at K=16

  // Per-fragment row base pointers (element offsets into bf16 rows).
  const uint16_t* arow[4];
  const uint16_t* brow[4];
#pragma unroll
  for (int i = 0; i < 4; ++i)
    arow[i] = Abf + (size_t)(bm + wm * 64 + i * 16 + lrow) * D_DIM + khalfA;
#pragma unroll
  for (int j = 0; j < 4; ++j)
    brow[j] = Bbf + (size_t)(bn + wn * 64 + j * 16 + lrow) * D_DIM + khalfB;

  v8f acc[4][4] = {};

  for (int k0 = 0; k0 < D_DIM; k0 += 32) {
    AFrag a[4], b[4];
#pragma unroll
    for (int i = 0; i < 4; ++i) {
      // A k order used here (same as B): lanes0-15 K0..15 contiguous; lanes16-31 K16..31
      a[i].q[0] = *(const uint4*)(arow[i] + k0);
      a[i].q[1] = *(const uint4*)(arow[i] + k0 + 8);
    }
#pragma unroll
    for (int j = 0; j < 4; ++j) {
      // B k order used here (same as A): lanes0-15 K0..15 contiguous; lanes16-31 K16..31
      b[j].q[0] = *(const uint4*)(brow[j] + k0);
      b[j].q[1] = *(const uint4*)(brow[j] + k0 + 8);
    }
#pragma unroll
    for (int i = 0; i < 4; ++i)
#pragma unroll
      for (int j = 0; j < 4; ++j)
        acc[i][j] = __builtin_amdgcn_wmma_f32_16x16x32_bf16(
            false, a[i].v, false, b[j].v, (short)0, acc[i][j], false, false);
  }

  // Stage t*dist tile into LDS. C layout: VGPR r, lanes0-15 -> M=r, N=lane;
  // lanes16-31 -> M=r+8, N=lane-16.
  const int rbase = (lane >> 4) * 8;
  const int ncol  = lane & 15;
#pragma unroll
  for (int i = 0; i < 4; ++i) {
#pragma unroll
    for (int j = 0; j < 4; ++j) {
      const int m0 = wm * 64 + i * 16 + rbase;
      const int n0 = wn * 64 + j * 16 + ncol;
#pragma unroll
      for (int r = 0; r < 8; ++r)
        lds[(size_t)(m0 + r) * LDS_STRIDE + n0] = acc[i][j][r] * 16.0f;
    }
  }

  __syncthreads();

  // Pooling: each output pools a 4x4 block of (t*dist).
  // right = sum_p log(sum_q e), left = sum_q log(sum_p e), out=(r+l)/128.
  const int ocol  = tid & 63;           // 64 output cols per block
  const int orow0 = (tid >> 6) * 8;     // 4 groups x 8 rows = 32 output rows
#pragma unroll
  for (int r = 0; r < 8; ++r) {
    const int orow = orow0 + r;
    float colsum[4] = {0.f, 0.f, 0.f, 0.f};
    float right = 0.f;
#pragma unroll
    for (int p = 0; p < 4; ++p) {
      const float* src = &lds[(size_t)(orow * 4 + p) * LDS_STRIDE + ocol * 4];
      float rs = 0.f;
#pragma unroll
      for (int q = 0; q < 4; ++q) {
        float e = __expf(src[q]);
        rs += e;
        colsum[q] += e;
      }
      right += __logf(rs);
    }
    float left = 0.f;
#pragma unroll
    for (int q = 0; q < 4; ++q) left += __logf(colsum[q]);
    const size_t og = (size_t)(blockIdx.y * 32 + orow) * BT_OUT
                    + (blockIdx.x * 64 + ocol);
    out[og] = (right + left) * (1.0f / 128.0f);
  }
}

// ------------------------------------------------------------------ launch ---
extern "C" void kernel_launch(void* const* d_in, const int* in_sizes, int n_in,
                              void* d_out, int out_size, void* d_ws, size_t ws_size,
                              hipStream_t stream) {
  const float* img = (const float*)d_in[0];   // [8192*1024] f32
  const float* txt = (const float*)d_in[1];   // [8192*1024] f32
  float* out = (float*)d_out;                 // [2048*2048] f32

  // Workspace: bf16 copies (16 MB + 16 MB).
  uint16_t* imgb = (uint16_t*)d_ws;
  uint16_t* txtb = imgb + (size_t)N_IMG * D_DIM;

  const int cvt_blocks = (N_IMG * D_DIM) / (256 * 8);   // 4096
  cvt_f32_bf16_kernel<<<cvt_blocks, 256, 0, stream>>>(img, imgb);
  cvt_f32_bf16_kernel<<<cvt_blocks, 256, 0, stream>>>(txt, txtb);

  dim3 grid(BT_OUT / 64, BI_OUT / 32);                  // (32, 64)
  size_t shmem = (size_t)BM * LDS_STRIDE * sizeof(float); // 133,120 B
  chamfer_gemm_kernel<<<grid, 256, shmem, stream>>>(imgb, txtb, out);
}
